// MMoEModel_26482768347290
// MI455X (gfx1250) — matrix operand order repaired and checked by reading, not compiled, not run on hardware
//
#include <hip/hip_runtime.h>
#include <math.h>

#define B_  4096
#define L_  64
#define DM_ 768
#define H_  4
#define E_  8
#define HD_ 384

typedef __attribute__((ext_vector_type(16))) __bf16 v16bf;
typedef __attribute__((ext_vector_type(8)))  __bf16 bf16x8;
typedef __attribute__((ext_vector_type(8)))  float  v8f;

// CDNA5 async global->LDS 16B copy (ASYNCcnt-tracked, no VGPR data path).
// Generic LDS pointer low 32 bits == LDS address (flat aperture truncation).
static __device__ __forceinline__ void async_g2l_b128(void* lds, const void* g) {
  unsigned loff = (unsigned)(unsigned long long)lds;
  unsigned long long ga = (unsigned long long)g;
  asm volatile("global_load_async_to_lds_b128 %0, %1, off"
               :: "v"(loff), "v"(ga) : "memory");
}
static __device__ __forceinline__ void async_wait0() {
  asm volatile("s_wait_asynccnt 0x0" ::: "memory");
}

// ---------------------------------------------------------------------------
// K0: mask-attention pooling.  One block per batch row; x[b] (64x768 f32,
// 192KB) staged in dynamic LDS via async DMA so HBM sees x exactly once.
// ---------------------------------------------------------------------------
__global__ void attn_pool_kernel(const float* __restrict__ x,
                                 const float* __restrict__ aw,
                                 float* __restrict__ feat,
                                 __bf16* __restrict__ featbf) {
  extern __shared__ char smem[];
  float* xs = (float*)smem;          // L_*DM_
  float* sc = xs + L_ * DM_;         // L_
  float* pr = sc + L_;               // L_
  const int tid = threadIdx.x;       // 256 threads = 8 waves
  const int b = blockIdx.x;

  // async-stage x[b] into LDS: 12288 16B chunks, 48 per thread
  {
    const char* src = (const char*)(x + (size_t)b * (L_ * DM_));
    char* dst = (char*)xs;
    for (int i = tid; i < (L_ * DM_ * 4) / 16; i += 256)
      async_g2l_b128(dst + (size_t)i * 16, src + (size_t)i * 16);
    async_wait0();
  }
  __syncthreads();

  const int w = tid >> 5, lane = tid & 31;
  float awr[DM_ / 32];
  #pragma unroll
  for (int j = 0; j < DM_ / 32; ++j) awr[j] = aw[lane + j * 32];

  // scores: wave w handles rows l = 8w .. 8w+7
  for (int j = 0; j < 8; ++j) {
    const int l = w * 8 + j;
    const float* r = xs + l * DM_;
    float s = 0.f, rs = 0.f;
    #pragma unroll
    for (int k = 0; k < DM_ / 32; ++k) {
      float v = r[lane + k * 32];
      s += v * awr[k];
      rs += v;
    }
    for (int off = 16; off > 0; off >>= 1) {
      s  += __shfl_xor(s,  off, 32);
      rs += __shfl_xor(rs, off, 32);
    }
    if (lane == 0) sc[l] = (rs != 0.f) ? s : -INFINITY;
  }
  __syncthreads();

  // softmax over L (cheap, redundant per thread)
  float mx = -INFINITY;
  for (int l = 0; l < L_; ++l) mx = fmaxf(mx, sc[l]);
  float den = 0.f;
  for (int l = 0; l < L_; ++l) den += __expf(sc[l] - mx);
  const float inv = 1.f / den;
  if (tid < L_) pr[tid] = __expf(sc[tid] - mx) * inv;
  __syncthreads();

  // feature = p . x
  #pragma unroll
  for (int j = 0; j < DM_ / 256; ++j) {
    const int d = tid + j * 256;
    float acc = 0.f;
    for (int l = 0; l < L_; ++l) acc += pr[l] * xs[l * DM_ + d];
    feat[(size_t)b * DM_ + d] = acc;
    featbf[(size_t)b * DM_ + d] = (__bf16)acc;   // hw cvt
  }
}

// ---------------------------------------------------------------------------
// Kp: bf16 copies of expert weights + folded BN scale/shift.
// ---------------------------------------------------------------------------
__global__ void prep_kernel(const float* __restrict__ eW1, const float* __restrict__ eW2,
                            const float* __restrict__ eb1, const float* __restrict__ bn1g,
                            const float* __restrict__ bn1b, const float* __restrict__ eb2,
                            const float* __restrict__ bn2g, const float* __restrict__ bn2b,
                            __bf16* __restrict__ W1bf, __bf16* __restrict__ W2bf,
                            float* __restrict__ s1, float* __restrict__ t1,
                            float* __restrict__ s2, float* __restrict__ t2) {
  const float inv = rsqrtf(1.0f + 1e-5f);
  size_t i = (size_t)blockIdx.x * blockDim.x + threadIdx.x;
  const size_t n1 = (size_t)E_ * DM_ * DM_;
  const size_t n2 = (size_t)E_ * DM_ * HD_;
  const size_t ns1 = (size_t)E_ * DM_;
  const size_t ns2 = (size_t)E_ * HD_;
  if (i < n1) { W1bf[i] = (__bf16)eW1[i]; return; }
  i -= n1;
  if (i < n2) { W2bf[i] = (__bf16)eW2[i]; return; }
  i -= n2;
  if (i < ns1) { float s = bn1g[i] * inv; s1[i] = s; t1[i] = eb1[i] * s + bn1b[i]; return; }
  i -= ns1;
  if (i < ns2) { float s = bn2g[i] * inv; s2[i] = s; t2[i] = eb2[i] * s + bn2b[i]; }
}

// ---------------------------------------------------------------------------
// K1: per-row gate for head D[b] only.
// ---------------------------------------------------------------------------
__global__ void gate_kernel(const float* __restrict__ feat, const int* __restrict__ D,
                            const float* __restrict__ gW1, const float* __restrict__ gb1,
                            const float* __restrict__ gW2, const float* __restrict__ gb2,
                            float* __restrict__ gate) {
  __shared__ float fsh[DM_];
  __shared__ float red[128 * E_];
  const int t = threadIdx.x;   // 128
  const int b = blockIdx.x;
  const int h = D[b];
  for (int i = t; i < DM_; i += 128) fsh[i] = feat[(size_t)b * DM_ + i];
  __syncthreads();

  const float* W1 = gW1 + (size_t)h * DM_ * HD_;
  float acc[3] = {0.f, 0.f, 0.f};
  for (int d = 0; d < DM_; ++d) {
    const float fd = fsh[d];
    const float* wr = W1 + (size_t)d * HD_;
    acc[0] += fd * wr[t];
    acc[1] += fd * wr[t + 128];
    acc[2] += fd * wr[t + 256];
  }
  float p2[E_];
  #pragma unroll
  for (int c = 0; c < E_; ++c) p2[c] = 0.f;
  #pragma unroll
  for (int j = 0; j < 3; ++j) {
    const int o = t + j * 128;
    const float hv = fmaxf(acc[j] + gb1[h * HD_ + o], 0.f);
    const float* w2 = gW2 + ((size_t)h * HD_ + o) * E_;
    #pragma unroll
    for (int c = 0; c < E_; ++c) p2[c] += hv * w2[c];
  }
  #pragma unroll
  for (int c = 0; c < E_; ++c) red[t * E_ + c] = p2[c];
  __syncthreads();
  for (int s = 64; s > 0; s >>= 1) {
    if (t < s)
      for (int c = 0; c < E_; ++c) red[t * E_ + c] += red[(t + s) * E_ + c];
    __syncthreads();
  }
  if (t == 0) {
    float lg[E_], m = -INFINITY;
    for (int c = 0; c < E_; ++c) { lg[c] = red[c] + gb2[h * E_ + c]; m = fmaxf(m, lg[c]); }
    float dn = 0.f;
    for (int c = 0; c < E_; ++c) dn += __expf(lg[c] - m);
    const float iv = 1.f / dn;
    for (int c = 0; c < E_; ++c) gate[(size_t)b * E_ + c] = __expf(lg[c] - m) * iv;
  }
}

// ---------------------------------------------------------------------------
// K2: fused expert MLP via v_wmma_f32_16x16x32_bf16, 2 N-tiles per pass so
// each A fragment (2x ds_load_b128) feeds two WMMAs.
// ---------------------------------------------------------------------------
__global__ void expert_kernel(const __bf16* __restrict__ featbf,
                              const __bf16* __restrict__ W1bf,
                              const __bf16* __restrict__ W2bf,
                              const float* __restrict__ s1, const float* __restrict__ t1,
                              const float* __restrict__ s2, const float* __restrict__ t2,
                              const float* __restrict__ gate,
                              float* __restrict__ h2w) {
  extern __shared__ char smem[];
  __bf16* Fs  = (__bf16*)smem;           // 32 x 768 bf16 (48KB)
  __bf16* C1  = Fs + 32 * DM_;           // 32 x 768 bf16 (48KB)
  float*  gsh = (float*)(C1 + 32 * DM_); // 32 gate values

  const int tid = threadIdx.x, w = tid >> 5, lane = tid & 31;
  const int tile = blockIdx.x, e = blockIdx.y;
  const int rowBase = tile * 32;

  // async-stage feature tile (bf16, 48KB): 3072 16B chunks, 12 per thread
  {
    const char* src = (const char*)(featbf + (size_t)rowBase * DM_);
    char* dst = (char*)Fs;
    for (int i = tid; i < (32 * DM_ * 2) / 16; i += 256)
      async_g2l_b128(dst + (size_t)i * 16, src + (size_t)i * 16);
    async_wait0();
  }
  if (tid < 32) gsh[tid] = gate[(size_t)(rowBase + tid) * E_ + e];
  __syncthreads();

  const int mBase = (w & 1) * 16;   // two 16-row m-tiles
  const int ng = w >> 1;            // four n-groups
  const int mA = mBase + (lane & 15);
  const int kAo = (lane >> 4) << 3;                 // 0 | 8
  const int kBo = (lane & 15) + ((lane >> 4) << 4); // lane's K row
  const int mo = (lane >> 4) << 3;                  // C-layout M offset

  // ---- stage 1: C1 = relu(s1*(F @ W1e) + t1) -> bf16 LDS ----
  const __bf16* W1 = W1bf + (size_t)e * DM_ * DM_;
  for (int tp = 0; tp < 6; ++tp) {
    const int nBase = (ng * 12 + tp * 2) * 16;
    v8f acc0 = {}, acc1 = {};
    #pragma unroll 4
    for (int kt = 0; kt < DM_ / 32; ++kt) {
      const int kBase = kt * 32;
      const __bf16* ap = Fs + mA * DM_ + kBase + kAo;
      union { bf16x8 h[2]; v16bf v; } ua;
      ua.h[0] = *(const bf16x8*)ap;
      ua.h[1] = *(const bf16x8*)(ap + 16);
      const __bf16* bp = W1 + (size_t)kBase * DM_ + nBase + (lane & 15);
      v16bf b0, b1;
      #pragma unroll
      for (int s = 0; s < 16; ++s) {
        const int kk = kAo + (s & 7) + ((s >> 3) << 4);
        b0[s] = bp[(size_t)kk * DM_];
        b1[s] = bp[(size_t)kk * DM_ + 16];
      }
      acc0 = __builtin_amdgcn_wmma_f32_16x16x32_bf16(false, ua.v, false, b0,
                                                     (short)0, acc0, false, false);
      acc1 = __builtin_amdgcn_wmma_f32_16x16x32_bf16(false, ua.v, false, b1,
                                                     (short)0, acc1, false, false);
    }
    const int ch0 = nBase + (lane & 15);
    const int ch1 = ch0 + 16;
    const float sc0 = s1[e * DM_ + ch0], sh0 = t1[e * DM_ + ch0];
    const float sc1 = s1[e * DM_ + ch1], sh1 = t1[e * DM_ + ch1];
    #pragma unroll
    for (int r = 0; r < 8; ++r) {
      const int m = mBase + mo + r;
      C1[m * DM_ + ch0] = (__bf16)fmaxf(acc0[r] * sc0 + sh0, 0.f);
      C1[m * DM_ + ch1] = (__bf16)fmaxf(acc1[r] * sc1 + sh1, 0.f);
    }
  }
  __syncthreads();

  // ---- stage 2: h2w[e,b,:] = gate[b,e] * relu(s2*(C1 @ W2e) + t2) ----
  const __bf16* W2 = W2bf + (size_t)e * DM_ * HD_;
  for (int tp = 0; tp < 3; ++tp) {
    const int nBase = (ng * 6 + tp * 2) * 16;
    v8f acc0 = {}, acc1 = {};
    #pragma unroll 4
    for (int kt = 0; kt < DM_ / 32; ++kt) {
      const int kBase = kt * 32;
      const __bf16* ap = C1 + mA * DM_ + kBase + kAo;
      union { bf16x8 h[2]; v16bf v; } ua;
      ua.h[0] = *(const bf16x8*)ap;
      ua.h[1] = *(const bf16x8*)(ap + 16);
      const __bf16* bp = W2 + (size_t)kBase * HD_ + nBase + (lane & 15);
      v16bf b0, b1;
      #pragma unroll
      for (int s = 0; s < 16; ++s) {
        const int kk = kAo + (s & 7) + ((s >> 3) << 4);
        b0[s] = bp[(size_t)kk * HD_];
        b1[s] = bp[(size_t)kk * HD_ + 16];
      }
      acc0 = __builtin_amdgcn_wmma_f32_16x16x32_bf16(false, ua.v, false, b0,
                                                     (short)0, acc0, false, false);
      acc1 = __builtin_amdgcn_wmma_f32_16x16x32_bf16(false, ua.v, false, b1,
                                                     (short)0, acc1, false, false);
    }
    const int ch0 = nBase + (lane & 15);
    const int ch1 = ch0 + 16;
    const float sc0 = s2[e * HD_ + ch0], sh0 = t2[e * HD_ + ch0];
    const float sc1 = s2[e * HD_ + ch1], sh1 = t2[e * HD_ + ch1];
    #pragma unroll
    for (int r = 0; r < 8; ++r) {
      const int ml = mBase + mo + r;
      const float gv = gsh[ml];
      float* orow = h2w + ((size_t)e * B_ + (rowBase + ml)) * HD_;
      orow[ch0] = fmaxf(acc0[r] * sc0 + sh0, 0.f) * gv;
      orow[ch1] = fmaxf(acc1[r] * sc1 + sh1, 0.f) * gv;
    }
  }
}

// ---------------------------------------------------------------------------
// K3: rep = sum_e h2w[e,b,:]; out = softmax(rep @ hW[D[b]] + hb[D[b]])
// ---------------------------------------------------------------------------
__global__ void head_kernel(const float* __restrict__ h2w, const int* __restrict__ D,
                            const float* __restrict__ hW, const float* __restrict__ hb,
                            float* __restrict__ out) {
  __shared__ float red[128 * 2];
  const int t = threadIdx.x, b = blockIdx.x;
  const int h = D[b];
  float p0 = 0.f, p1 = 0.f;
  #pragma unroll
  for (int j = 0; j < 3; ++j) {
    const int o = t + j * 128;
    float r = 0.f;
    #pragma unroll
    for (int e = 0; e < E_; ++e) r += h2w[((size_t)e * B_ + b) * HD_ + o];
    const float* wrow = hW + ((size_t)h * HD_ + o) * 2;
    p0 += r * wrow[0];
    p1 += r * wrow[1];
  }
  red[t * 2] = p0; red[t * 2 + 1] = p1;
  __syncthreads();
  for (int s = 64; s > 0; s >>= 1) {
    if (t < s) { red[t * 2] += red[(t + s) * 2]; red[t * 2 + 1] += red[(t + s) * 2 + 1]; }
    __syncthreads();
  }
  if (t == 0) {
    const float l0 = red[0] + hb[h * 2];
    const float l1 = red[1] + hb[h * 2 + 1];
    const float m = fmaxf(l0, l1);
    const float e0 = __expf(l0 - m), e1 = __expf(l1 - m);
    const float iv = 1.f / (e0 + e1);
    out[(size_t)b * 2]     = e0 * iv;
    out[(size_t)b * 2 + 1] = e1 * iv;
  }
}

// ---------------------------------------------------------------------------
extern "C" void kernel_launch(void* const* d_in, const int* in_sizes, int n_in,
                              void* d_out, int out_size, void* d_ws, size_t ws_size,
                              hipStream_t stream) {
  const float* x    = (const float*)d_in[0];
  const int*   D    = (const int*)  d_in[1];
  const float* aw   = (const float*)d_in[2];
  const float* gW1  = (const float*)d_in[3];
  const float* gb1  = (const float*)d_in[4];
  const float* gW2  = (const float*)d_in[5];
  const float* gb2  = (const float*)d_in[6];
  const float* eW1  = (const float*)d_in[7];
  const float* eb1  = (const float*)d_in[8];
  const float* bn1g = (const float*)d_in[9];
  const float* bn1b = (const float*)d_in[10];
  const float* eW2  = (const float*)d_in[11];
  const float* eb2  = (const float*)d_in[12];
  const float* bn2g = (const float*)d_in[13];
  const float* bn2b = (const float*)d_in[14];
  const float* hW   = (const float*)d_in[15];
  const float* hb   = (const float*)d_in[16];
  float* out = (float*)d_out;
  (void)in_sizes; (void)n_in; (void)out_size; (void)ws_size;

  char* wsp = (char*)d_ws;
  size_t off = 0;
  auto take = [&](size_t bytes) -> char* {
    char* p = wsp + off;
    off = (off + bytes + 255) & ~(size_t)255;
    return p;
  };
  float*  feat   = (float*)  take((size_t)B_ * DM_ * 4);
  __bf16* featbf = (__bf16*) take((size_t)B_ * DM_ * 2);
  float*  gate   = (float*)  take((size_t)B_ * E_ * 4);
  __bf16* W1bf   = (__bf16*) take((size_t)E_ * DM_ * DM_ * 2);
  __bf16* W2bf   = (__bf16*) take((size_t)E_ * DM_ * HD_ * 2);
  float*  s1     = (float*)  take((size_t)E_ * DM_ * 4);
  float*  t1     = (float*)  take((size_t)E_ * DM_ * 4);
  float*  s2     = (float*)  take((size_t)E_ * HD_ * 4);
  float*  t2     = (float*)  take((size_t)E_ * HD_ * 4);
  float*  h2w    = (float*)  take((size_t)E_ * B_ * HD_ * 4);

  const size_t smem0 = (size_t)(L_ * DM_ + 2 * L_) * 4;     // ~193KB (<320KB WGP LDS)
  const size_t smem2 = (size_t)32 * DM_ * 2 * 2 + 32 * 4;   // ~96.1KB
  (void)hipFuncSetAttribute(reinterpret_cast<const void*>(attn_pool_kernel),
                            hipFuncAttributeMaxDynamicSharedMemorySize, (int)smem0);
  (void)hipFuncSetAttribute(reinterpret_cast<const void*>(expert_kernel),
                            hipFuncAttributeMaxDynamicSharedMemorySize, (int)smem2);

  hipLaunchKernelGGL(attn_pool_kernel, dim3(B_), dim3(256), smem0, stream,
                     x, aw, feat, featbf);

  const size_t ptot = (size_t)E_ * DM_ * DM_ + (size_t)E_ * DM_ * HD_
                    + (size_t)E_ * DM_ + (size_t)E_ * HD_;
  hipLaunchKernelGGL(prep_kernel, dim3((unsigned)((ptot + 255) / 256)), dim3(256), 0, stream,
                     eW1, eW2, eb1, bn1g, bn1b, eb2, bn2g, bn2b,
                     W1bf, W2bf, s1, t1, s2, t2);

  hipLaunchKernelGGL(gate_kernel, dim3(B_), dim3(128), 0, stream,
                     feat, D, gW1, gb1, gW2, gb2, gate);

  hipLaunchKernelGGL(expert_kernel, dim3(B_ / 32, E_), dim3(256), smem2, stream,
                     featbf, W1bf, W2bf, s1, t1, s2, t2, gate, h2w);

  hipLaunchKernelGGL(head_kernel, dim3(B_), dim3(128), 0, stream,
                     h2w, D, hW, hb, out);
}
